// skill_net_17858474016960
// MI455X (gfx1250) — matrix operand order repaired and checked by reading, not compiled, not run on hardware
//
#include <hip/hip_runtime.h>

// ---------------------------------------------------------------------------
// Fused 3-layer noisy-net ensemble MLP for MI455X (gfx1250, wave32, WMMA).
//   Pre-pass: convert fp32 weights -> f16 once into d_ws (reused 128x via L2).
//   Main: one WG per (32-row batch tile, model). All GEMMs via
//   v_wmma_f32_16x16x32_f16 (f32 accumulate); activations stay in LDS.
//   Register blocking: each wave owns a 32x32 output super-tile -> per k-step
//   2 A-frags + 4 B-frags feed 8 WMMAs (1.5 loads per WMMA).
// ---------------------------------------------------------------------------

typedef __attribute__((ext_vector_type(16))) _Float16 v16h;
typedef __attribute__((ext_vector_type(8)))  _Float16 v8h;
typedef __attribute__((ext_vector_type(4)))  _Float16 v4h;
typedef __attribute__((ext_vector_type(8)))  float    v8f;

#define N_M     16
#define IN_DIM  256
#define HID     512
#define OUT_DIM 64
#define TILE_M  32

// LDS layout (units: halves). Row strides padded +8 halves (16B) so the 16
// lanes' ds_load_b128 for an A-fragment hit distinct bank quads.
#define XIN_STRIDE 264
#define X_STRIDE   520
#define X1_OFF  (TILE_M * XIN_STRIDE)
#define X2_OFF  (X1_OFF + TILE_M * X_STRIDE)
#define FLT_OFF (X2_OFF + TILE_M * X_STRIDE)          // even -> float aligned
#define SMEM_HALVES (FLT_OFF + 2 * (HID + HID + TILE_M + TILE_M))
#define SMEM_BYTES  (SMEM_HALVES * 2)                 // = 87,808 bytes

// f16 weight offsets inside d_ws (units: halves)
#define OFF_W11 ((size_t)0)
#define OFF_W12 ((size_t)2097152)
#define OFF_W21 ((size_t)4194304)
#define OFF_W22 ((size_t)8388608)
#define OFF_W31 ((size_t)12582912)
#define OFF_W32 ((size_t)13107200)

#define WMMA_F16(a, b, c) \
    __builtin_amdgcn_wmma_f32_16x16x32_f16(false, (a), false, (b), (short)0, \
                                           (c), false, false)

__device__ __forceinline__ float sgn_sqrt(float x) {
    return copysignf(sqrtf(fabsf(x)), x);   // sign(x)*sqrt(|x|)
}

// A fragment (16x32 f16, M x K): lane&15 = M row; element i holds
// K = k0 + 16*(lane>=16) + i  -> two contiguous v8h loads.
__device__ __forceinline__ v16h load_a_frag(const _Float16* X, int stride,
                                            int mbase, int lane, int k0) {
    const _Float16* p = X + (mbase + (lane & 15)) * stride + k0 + 16 * (lane >> 4);
    v8h lo = *(const v8h*)p;
    v8h hi = *(const v8h*)(p + 8);
    return __builtin_shufflevector(lo, hi, 0,1,2,3,4,5,6,7,8,9,10,11,12,13,14,15);
}

// B fragment (32x16 f16, K x N) built from row-major W[out][in] (B = W^T):
// lane&15 = N column (= W row), K = k0 + 16*(lane>=16) + i  -> 16 contiguous
// halves per lane = two global_load_b128.
__device__ __forceinline__ v16h load_b_frag(const _Float16* W, int Kdim,
                                            int nbase, int lane, int k0) {
    const _Float16* p = W + (size_t)(nbase + (lane & 15)) * Kdim
                          + k0 + 16 * (lane >> 4);
    v8h lo = *(const v8h*)p;
    v8h hi = *(const v8h*)(p + 8);
    return __builtin_shufflevector(lo, hi, 0,1,2,3,4,5,6,7,8,9,10,11,12,13,14,15);
}

// Hidden layer: Y = relu((X@Wmu^T + bmu) + gi*gj*exp(clip(X@Wls^T + bls)))
// Register-blocked: each wave computes a 32x32 super-tile of both outputs.
template<int K, int N>
__device__ __forceinline__ void hidden_layer(
    const _Float16* __restrict__ Xl, int sin,
    const _Float16* __restrict__ wmu, const _Float16* __restrict__ wls,
    const float* __restrict__ bmu, const float* __restrict__ bls,
    const float* __restrict__ gi, const float* __restrict__ gj,
    _Float16* __restrict__ Xo, int sout, int wave, int lane)
{
    const int NSUP = N / 32;                          // 32-wide n super-tiles
    const int half = lane >> 4, l15 = lane & 15;
    for (int sp = wave; sp < NSUP; sp += 8) {         // wave-uniform: EXEC all-1s
        const int n0 = sp * 32;
        v8f am[2][2] = {};                            // [mt][ntl]
        v8f al[2][2] = {};
        #pragma unroll 2
        for (int kt = 0; kt < K / 32; ++kt) {
            const int k0 = kt * 32;
            // WGP-scope prefetch (locality 3) of next k-chunk: 32 lanes cover
            // the 32 weight rows of this super-tile, one line ahead of use.
            if (kt + 1 < K / 32) {
                const size_t pre = (size_t)(n0 + lane) * K + k0 + 32;
                __builtin_prefetch(wmu + pre, 0, 3);
                __builtin_prefetch(wls + pre, 0, 3);
            }
            v16h a0 = load_a_frag(Xl, sin, 0,  lane, k0);
            v16h a1 = load_a_frag(Xl, sin, 16, lane, k0);
            v16h bm0 = load_b_frag(wmu, K, n0,      lane, k0);
            v16h bm1 = load_b_frag(wmu, K, n0 + 16, lane, k0);
            v16h bl0 = load_b_frag(wls, K, n0,      lane, k0);
            v16h bl1 = load_b_frag(wls, K, n0 + 16, lane, k0);
            am[0][0] = WMMA_F16(a0, bm0, am[0][0]);
            am[1][0] = WMMA_F16(a1, bm0, am[1][0]);
            am[0][1] = WMMA_F16(a0, bm1, am[0][1]);
            am[1][1] = WMMA_F16(a1, bm1, am[1][1]);
            al[0][0] = WMMA_F16(a0, bl0, al[0][0]);
            al[1][0] = WMMA_F16(a1, bl0, al[1][0]);
            al[0][1] = WMMA_F16(a0, bl1, al[0][1]);
            al[1][1] = WMMA_F16(a1, bl1, al[1][1]);
        }
        #pragma unroll
        for (int ntl = 0; ntl < 2; ++ntl) {
            const int   ng  = n0 + ntl * 16 + l15;
            const float bm  = bmu[ng];
            const float bl  = bls[ng];
            const float gjv = gj[ng];
            #pragma unroll
            for (int mt = 0; mt < 2; ++mt) {
                #pragma unroll
                for (int r = 0; r < 8; ++r) {         // C/D: M = r + 8*half
                    const int m  = mt * 16 + r + 8 * half;
                    const float mu = am[mt][ntl][r] + bm;
                    const float ls = fminf(fmaxf(al[mt][ntl][r] + bl, -20.f), 4.f);
                    const float x  = fmaxf(mu + gi[m] * gjv * __expf(ls), 0.f);
                    Xo[m * sout + ng] = (_Float16)x;
                }
            }
        }
    }
}

// Output layer: m = X@W31^T + b31 ; log_stdev = clip(X@W32^T + b32, -20, 2)
template<int K>
__device__ __forceinline__ void final_layer(
    const _Float16* __restrict__ Xl, int sin,
    const _Float16* __restrict__ wm, const _Float16* __restrict__ wl,
    const float* __restrict__ bm_, const float* __restrict__ bl_,
    float* __restrict__ out, int Btot, int b0, int j, int wave, int lane)
{
    const int NPOS = (TILE_M / 16) * (OUT_DIM / 16);  // 8 -> one per wave
    const int half = lane >> 4, l15 = lane & 15;
    const size_t ls_off = (size_t)Btot * N_M * OUT_DIM;
    for (int pos = wave; pos < NPOS; pos += 8) {
        const int mt = pos & 1, nt = pos >> 1;
        v8f am = {};
        v8f al = {};
        #pragma unroll 4
        for (int kt = 0; kt < K / 32; ++kt) {
            const int k0 = kt * 32;
            v16h a  = load_a_frag(Xl, sin, mt * 16, lane, k0);
            v16h b0 = load_b_frag(wm, K, nt * 16, lane, k0);
            v16h b1 = load_b_frag(wl, K, nt * 16, lane, k0);
            am = WMMA_F16(a, b0, am);
            al = WMMA_F16(a, b1, al);
        }
        const int   ng = nt * 16 + l15;
        const float bm = bm_[ng];
        const float bl = bl_[ng];
        #pragma unroll
        for (int r = 0; r < 8; ++r) {
            const int m = mt * 16 + r + 8 * half;
            const size_t o = ((size_t)(b0 + m) * N_M + j) * OUT_DIM + ng;
            out[o]          = am[r] + bm;
            out[ls_off + o] = fminf(fmaxf(al[r] + bl, -20.f), 2.f);
        }
    }
}

__global__ void cvt_f32_to_f16(const float* __restrict__ src,
                               _Float16* __restrict__ dst, int n4) {
    int i = blockIdx.x * blockDim.x + threadIdx.x;
    if (i < n4) {
        float4 v = *(const float4*)(src + (size_t)i * 4);
        v4h h;
        h[0] = (_Float16)v.x; h[1] = (_Float16)v.y;
        h[2] = (_Float16)v.z; h[3] = (_Float16)v.w;
        *(v4h*)(dst + (size_t)i * 4) = h;
    }
}

__global__ __launch_bounds__(256) void fused_noisy_mlp(
    const float* __restrict__ s,
    const float* __restrict__ ei1, const float* __restrict__ ej1,
    const float* __restrict__ ei2, const float* __restrict__ ej2,
    const float* __restrict__ b11, const float* __restrict__ b12,
    const float* __restrict__ b21, const float* __restrict__ b22,
    const float* __restrict__ b31, const float* __restrict__ b32,
    const _Float16* __restrict__ w11h, const _Float16* __restrict__ w12h,
    const _Float16* __restrict__ w21h, const _Float16* __restrict__ w22h,
    const _Float16* __restrict__ w31h, const _Float16* __restrict__ w32h,
    float* __restrict__ out, int Btot)
{
    extern __shared__ _Float16 smem[];
    _Float16* Xin = smem;
    _Float16* X1  = smem + X1_OFF;
    _Float16* X2  = smem + X2_OFF;
    float* gj1 = (float*)(smem + FLT_OFF);
    float* gj2 = gj1 + HID;
    float* gi1 = gj2 + HID;
    float* gi2 = gi1 + TILE_M;

    const int tid  = threadIdx.x;
    const int wave = tid >> 5;
    const int lane = tid & 31;
    const int b0   = blockIdx.x * TILE_M;
    const int j    = blockIdx.y;

    // Stage s tile (32x256 f32) -> f16 LDS; 8 float4 loads per thread.
    for (int e = tid; e < TILE_M * IN_DIM / 4; e += 256) {
        const int idx4 = e * 4;
        const int row  = idx4 >> 8;            // / IN_DIM
        const int col  = idx4 & (IN_DIM - 1);
        float4 v = *(const float4*)(s + ((size_t)(b0 + row) * N_M + j) * IN_DIM + col);
        v4h h;
        h[0] = (_Float16)v.x; h[1] = (_Float16)v.y;
        h[2] = (_Float16)v.z; h[3] = (_Float16)v.w;
        *(v4h*)(&Xin[row * XIN_STRIDE + col]) = h;
    }
    // Separable noise factors: noise[b][n] = gi[b] * gj[n].
    if (tid < TILE_M)
        gi1[tid] = sgn_sqrt(ei1[(size_t)(b0 + tid) * N_M + j]);
    else if (tid < 2 * TILE_M)
        gi2[tid - TILE_M] = sgn_sqrt(ei2[(size_t)(b0 + tid - TILE_M) * N_M + j]);
    for (int n = tid; n < HID; n += 256) {
        gj1[n] = sgn_sqrt(ej1[(size_t)j * HID + n]);
        gj2[n] = sgn_sqrt(ej2[(size_t)j * HID + n]);
    }
    __syncthreads();

    hidden_layer<IN_DIM, HID>(Xin, XIN_STRIDE,
                              w11h + (size_t)j * HID * IN_DIM,
                              w12h + (size_t)j * HID * IN_DIM,
                              b11 + (size_t)j * HID, b12 + (size_t)j * HID,
                              gi1, gj1, X1, X_STRIDE, wave, lane);
    __syncthreads();
    hidden_layer<HID, HID>(X1, X_STRIDE,
                           w21h + (size_t)j * HID * HID,
                           w22h + (size_t)j * HID * HID,
                           b21 + (size_t)j * HID, b22 + (size_t)j * HID,
                           gi2, gj2, X2, X_STRIDE, wave, lane);
    __syncthreads();
    final_layer<HID>(X2, X_STRIDE,
                     w31h + (size_t)j * OUT_DIM * HID,
                     w32h + (size_t)j * OUT_DIM * HID,
                     b31 + (size_t)j * OUT_DIM, b32 + (size_t)j * OUT_DIM,
                     out, Btot, b0, j, wave, lane);
}

extern "C" void kernel_launch(void* const* d_in, const int* in_sizes, int n_in,
                              void* d_out, int out_size, void* d_ws, size_t ws_size,
                              hipStream_t stream) {
    const float* s   = (const float*)d_in[0];
    const float* ei1 = (const float*)d_in[1];
    const float* ej1 = (const float*)d_in[2];
    const float* ei2 = (const float*)d_in[3];
    const float* ej2 = (const float*)d_in[4];
    const float* w11 = (const float*)d_in[5];
    const float* b11 = (const float*)d_in[6];
    const float* w12 = (const float*)d_in[7];
    const float* b12 = (const float*)d_in[8];
    const float* w21 = (const float*)d_in[9];
    const float* b21 = (const float*)d_in[10];
    const float* w22 = (const float*)d_in[11];
    const float* b22 = (const float*)d_in[12];
    const float* w31 = (const float*)d_in[13];
    const float* b31 = (const float*)d_in[14];
    const float* w32 = (const float*)d_in[15];
    const float* b32 = (const float*)d_in[16];

    const int Btot = in_sizes[0] / (N_M * IN_DIM);   // 4096
    _Float16* ws = (_Float16*)d_ws;

    // Pre-pass: fp32 -> f16 weight conversion (weights then reused
    // Btot/TILE_M = 128x per model out of the 192MB L2).
    struct Cvt { const float* src; size_t off; int n; };
    const Cvt cv[6] = {
        { w11, OFF_W11, N_M * HID * IN_DIM },
        { w12, OFF_W12, N_M * HID * IN_DIM },
        { w21, OFF_W21, N_M * HID * HID },
        { w22, OFF_W22, N_M * HID * HID },
        { w31, OFF_W31, N_M * OUT_DIM * HID },
        { w32, OFF_W32, N_M * OUT_DIM * HID },
    };
    for (int i = 0; i < 6; ++i) {
        const int n4 = cv[i].n / 4;
        cvt_f32_to_f16<<<(n4 + 255) / 256, 256, 0, stream>>>(cv[i].src,
                                                             ws + cv[i].off, n4);
    }

    dim3 grid(Btot / TILE_M, N_M);
    fused_noisy_mlp<<<grid, 256, SMEM_BYTES, stream>>>(
        s, ei1, ej1, ei2, ej2,
        b11, b12, b21, b22, b31, b32,
        ws + OFF_W11, ws + OFF_W12, ws + OFF_W21,
        ws + OFF_W22, ws + OFF_W31, ws + OFF_W32,
        (float*)d_out, Btot);
}
